// EmbSeqEncoder_35785667510358
// MI455X (gfx1250) — matrix operand order repaired and checked by reading, not compiled, not run on hardware
//
#include <hip/hip_runtime.h>
#include <hip/hip_bf16.h>

// CDNA5 (gfx1250) wave32 WMMA types
typedef __attribute__((ext_vector_type(16))) __bf16 v16bf;
typedef __attribute__((ext_vector_type(8)))  float  v8f;
typedef __attribute__((ext_vector_type(4)))  float  v4f;
typedef __attribute__((ext_vector_type(2)))  float  v2f;

#define D_DIM   256   // output feature dim (N)
#define K_DIM   768   // reduction dim
#define KCH     24    // K chunks of 32
#define NTILES  16    // 256/16 col tiles
// packed W fragment: [nt][kc][lane][16 bf16] -> (nt*KCH + kc)*512 + lane*16 elems
#define WPACK_ELEMS (NTILES * KCH * 32 * 16)   // 196608 bf16 per (hi|lo)

union BF2 { __bf16 h[2]; unsigned int u; };

__device__ __forceinline__ __bf16 f2bf(float f) { return (__bf16)f; }
__device__ __forceinline__ float  bf2f(__bf16 h) { return (float)h; }

// ---------------------------------------------------------------------------
// Kernel 1: split W (f32, row-major [K][N]) into bf16 hi/lo, pre-packed into
// per-lane WMMA B-fragment order for V_WMMA_F32_16X16X32_BF16.
// B (32x16, KxN), this file's k order: lane = half*16 + n; VGPR v packs k = half*16 + 2v + pos.
// ---------------------------------------------------------------------------
__global__ void EmbSeq_wsplit_pack(const float* __restrict__ W,
                                   __bf16* __restrict__ whi,
                                   __bf16* __restrict__ wlo) {
    int idx = blockIdx.x * blockDim.x + threadIdx.x;
    if (idx >= WPACK_ELEMS) return;
    int i    = idx & 15;          // element within lane fragment
    int lane = (idx >> 4) & 31;
    int rest = idx >> 9;          // nt*KCH + kc
    int kc   = rest % KCH;
    int nt   = rest / KCH;
    int half = lane >> 4;
    int n    = lane & 15;
    int v    = i >> 1;
    int pos  = i & 1;
    int klocal = half * 16 + 2 * v + pos;      // 0..31
    int k   = kc * 32 + klocal;
    int col = nt * 16 + n;
    float w = W[k * D_DIM + col];
    __bf16 h = f2bf(w);
    whi[idx] = h;
    wlo[idx] = f2bf(w - bf2f(h));
}

// ---------------------------------------------------------------------------
// A-chunk stage: thread t owns one K-pair of the 16x32 f32 chunk.
// Fragment layout used here (A 16x32, same k order as B): lane = half*16 + m;
// elem i: v=i>>1,pos=i&1; k = half*16 + v*2 + pos.
// Pair p = lane*8 + v  ->  one b64 global load, one b32 LDS store per hi/lo.
// ---------------------------------------------------------------------------
__device__ __forceinline__ void stage_chunk(const float* __restrict__ A,
                                            long t0, int nrows_total, int kc,
                                            unsigned int* __restrict__ dhi,
                                            unsigned int* __restrict__ dlo,
                                            int t) {
    const int p  = t;                 // pair index 0..255
    const int sl = p >> 3;            // destination lane
    const int j  = p & 7;             // fragment VGPR index v
    const int sm = sl & 15;
    const int sh = sl >> 4;
    const int kb = sh * 16 + (j << 1);   // even, 0..30
    long srow = t0 + sm;
    if (srow >= nrows_total) srow = nrows_total - 1;            // clamp (robustness)
    const v2f a2 = __builtin_nontemporal_load(
        (const v2f*)(A + srow * K_DIM + kc * 32 + kb));         // streamed once: NT
    BF2 hh, ll;
    hh.h[0] = f2bf(a2.x);
    hh.h[1] = f2bf(a2.y);
    ll.h[0] = f2bf(a2.x - bf2f(hh.h[0]));
    ll.h[1] = f2bf(a2.y - bf2f(hh.h[1]));
    dhi[p] = hh.u;
    dlo[p] = ll.u;
}

// ---------------------------------------------------------------------------
// Kernel 2: fused GEMM (split-bf16, 3-product ~f32 accuracy) + ragged->padded
// scatter. One block per 16-row padded tile; 8 waves x 2 N-tiles each.
// Double-buffered LDS A stage: one barrier per K-chunk, global loads of the
// next chunk overlap the 6 WMMAs of the current chunk.
// ---------------------------------------------------------------------------
__global__ __launch_bounds__(256)
void EmbSeqEncoder_gemm_scatter(const float* __restrict__ A,
                                const __bf16* __restrict__ Whi,
                                const __bf16* __restrict__ Wlo,
                                const float* __restrict__ bias,
                                const int*   __restrict__ lengths,
                                int max_len, int nrows_total,
                                float* __restrict__ out) {
    __shared__ __align__(32) __bf16 sAhi[2][512];
    __shared__ __align__(32) __bf16 sAlo[2][512];

    const int t  = threadIdx.x;
    const int p0 = blockIdx.x * 16;               // padded row base
    const int batch   = p0 / max_len;
    const int within0 = p0 - batch * max_len;

    // uniform (scalar) prefix sum over <=16 lengths
    int offs = 0;
    for (int j = 0; j < batch; ++j) offs += lengths[j];
    const int len = lengths[batch];

    float* outp = out + (long)p0 * D_DIM;

    if (within0 >= len) {
        // pure padding tile: stream zeros (each row written exactly once)
        const int row  = t >> 4;
        const int col0 = (t & 15) * 16;
        v4f z = {0.f, 0.f, 0.f, 0.f};
        v4f* rp = (v4f*)(outp + row * D_DIM + col0);
        __builtin_nontemporal_store(z, rp + 0);
        __builtin_nontemporal_store(z, rp + 1);
        __builtin_nontemporal_store(z, rp + 2);
        __builtin_nontemporal_store(z, rp + 3);
        return;
    }

    const long t0   = (long)offs + within0;       // source row base in A
    const int wave  = t >> 5;
    const int lane  = t & 31;
    const int half  = lane >> 4;
    const int m     = lane & 15;
    const int n0    = wave * 32;                  // this wave's 32 output cols

    v8f acc0 = {};
    v8f acc1 = {};

    const v16bf* __restrict__ WH = (const v16bf*)Whi;   // one v16bf == one lane fragment
    const v16bf* __restrict__ WL = (const v16bf*)Wlo;
    const long fragBase0 = (long)(wave * 2 + 0) * KCH * 32;  // in v16bf units
    const long fragBase1 = (long)(wave * 2 + 1) * KCH * 32;

    // prologue: stage chunk 0 into buffer 0
    stage_chunk(A, t0, nrows_total, 0, (unsigned int*)sAhi[0], (unsigned int*)sAlo[0], t);
    __syncthreads();

    for (int kc = 0; kc < KCH; ++kc) {
        const int cur = kc & 1;

        const v16bf ahi = *(const v16bf*)(&sAhi[cur][lane * 16]);   // ds_load_b128 x2
        const v16bf alo = *(const v16bf*)(&sAlo[cur][lane * 16]);

        // overlap: stage next chunk into the other buffer while WMMAs run
        if (kc + 1 < KCH) {
            stage_chunk(A, t0, nrows_total, kc + 1,
                        (unsigned int*)sAhi[cur ^ 1], (unsigned int*)sAlo[cur ^ 1], t);
        }

        const v16bf bh0 = WH[fragBase0 + kc * 32 + lane];
        const v16bf bl0 = WL[fragBase0 + kc * 32 + lane];
        const v16bf bh1 = WH[fragBase1 + kc * 32 + lane];
        const v16bf bl1 = WL[fragBase1 + kc * 32 + lane];

        // C += Ahi*Bhi + Ahi*Blo + Alo*Bhi, alternating accumulators
        acc0 = __builtin_amdgcn_wmma_f32_16x16x32_bf16(false, ahi, false, bh0,
                                                       (short)0, acc0, false, false);
        acc1 = __builtin_amdgcn_wmma_f32_16x16x32_bf16(false, ahi, false, bh1,
                                                       (short)0, acc1, false, false);
        acc0 = __builtin_amdgcn_wmma_f32_16x16x32_bf16(false, ahi, false, bl0,
                                                       (short)0, acc0, false, false);
        acc1 = __builtin_amdgcn_wmma_f32_16x16x32_bf16(false, ahi, false, bl1,
                                                       (short)0, acc1, false, false);
        acc0 = __builtin_amdgcn_wmma_f32_16x16x32_bf16(false, alo, false, bh0,
                                                       (short)0, acc0, false, false);
        acc1 = __builtin_amdgcn_wmma_f32_16x16x32_bf16(false, alo, false, bh1,
                                                       (short)0, acc1, false, false);

        __syncthreads();   // handoff: next-chunk stores visible, this buffer reusable
    }

    // --- bias + scatter-store; D layout: VGPR r -> row m = r + 8*half, col = lane&15
    const float bv0 = bias[n0 + m];
    const float bv1 = bias[n0 + 16 + m];
    #pragma unroll
    for (int r = 0; r < 8; ++r) {
        const int row = r + half * 8;
        const bool live = (within0 + row) < len;   // uniform here; robust for odd lens
        const float o0 = live ? (acc0[r] + bv0) : 0.f;
        const float o1 = live ? (acc1[r] + bv1) : 0.f;
        __builtin_nontemporal_store(o0, outp + (long)row * D_DIM + n0 + m);
        __builtin_nontemporal_store(o1, outp + (long)row * D_DIM + n0 + 16 + m);
    }
}

// ---------------------------------------------------------------------------
extern "C" void kernel_launch(void* const* d_in, const int* in_sizes, int n_in,
                              void* d_out, int out_size, void* d_ws, size_t ws_size,
                              hipStream_t stream) {
    const float* sent    = (const float*)d_in[0];   // [T, 768] f32
    const float* W       = (const float*)d_in[1];   // [768, 256] f32
    const float* b       = (const float*)d_in[2];   // [256] f32
    const int*   lengths = (const int*)d_in[3];     // [B] int32
    float*       out     = (float*)d_out;           // [B*max_len*256] f32

    const int B_docs  = in_sizes[3];
    const int T_rows  = in_sizes[0] / K_DIM;
    const int max_len = out_size / (B_docs * D_DIM);   // derived, no device read needed
    const int m_tiles = (B_docs * max_len) / 16;

    __bf16* whi = (__bf16*)d_ws;
    __bf16* wlo = whi + WPACK_ELEMS;

    // 1) split + pack W (runs once per launch; cheap, L2-resident afterwards)
    EmbSeq_wsplit_pack<<<(WPACK_ELEMS + 255) / 256, 256, 0, stream>>>(W, whi, wlo);

    // 2) fused GEMM + ragged->padded scatter
    EmbSeqEncoder_gemm_scatter<<<m_tiles, 256, 0, stream>>>(
        sent, whi, wlo, b, lengths, max_len, T_rows, out);
}
